// SDSABlock_46789373722895
// MI455X (gfx1250) — matrix operand order repaired and checked by reading, not compiled, not run on hardware
//
#include <hip/hip_runtime.h>
#include <hip/hip_bf16.h>

typedef _Float16 half_t;
typedef __attribute__((ext_vector_type(16))) _Float16 v16h;
typedef __attribute__((ext_vector_type(8)))  _Float16 v8h;
typedef __attribute__((ext_vector_type(8)))  float    v8f;

union Frag {
    v16h f;
    v8h  h[2];
};

// Problem constants (fixed by the reference harness)
constexpr int Tn  = 4;
constexpr int Bn  = 32;
constexpr int Cn  = 512;
constexpr int HWn = 196;                 // 14*14
constexpr int Mn  = Tn * Bn * HWn;       // 25088 GEMM rows
constexpr float BN_EPS = 1e-5f;

// ---------------------------------------------------------------------------
// K0: LIF over input x [T,B,C,HW] (f32) -> spike matrix xs [T*B*HW, C] (f16)
// (tau=2, v_th=1, hard reset to 0)
// ---------------------------------------------------------------------------
__global__ void lif_input_kernel(const float* __restrict__ x,
                                 half_t* __restrict__ xs) {
    int tid = blockIdx.x * blockDim.x + threadIdx.x;
    if (tid >= Bn * HWn * Cn) return;
    int c  = tid & (Cn - 1);
    int r  = tid >> 9;            // / Cn
    int hw = r % HWn;
    int b  = r / HWn;
    float v = 0.0f;
    #pragma unroll
    for (int t = 0; t < Tn; ++t) {
        float xt = x[((size_t)((t * Bn + b) * Cn + c)) * HWn + hw];
        v = 0.5f * (v + xt);                       // v += (x - v)/tau
        float s = (v >= 1.0f) ? 1.0f : 0.0f;
        v = (s > 0.0f) ? 0.0f : v;                 // hard reset
        xs[((size_t)((t * Bn + b) * HWn + hw)) * Cn + c] = (half_t)s;
    }
}

// ---------------------------------------------------------------------------
// K1: fold BN scale into f16 weights, build fused [1536,512] qkv weight and
// [512,512] out weight; per-output-channel additive shifts.
// ---------------------------------------------------------------------------
__global__ void prep_weights_kernel(
    const float* __restrict__ q_w, const float* __restrict__ k_w,
    const float* __restrict__ v_w, const float* __restrict__ o_w,
    const float* __restrict__ o_bias,
    const float* __restrict__ q_g, const float* __restrict__ q_b,
    const float* __restrict__ q_m, const float* __restrict__ q_v,
    const float* __restrict__ k_g, const float* __restrict__ k_b,
    const float* __restrict__ k_m, const float* __restrict__ k_v,
    const float* __restrict__ v_g, const float* __restrict__ v_b,
    const float* __restrict__ v_m, const float* __restrict__ v_v,
    const float* __restrict__ o_g, const float* __restrict__ o_beta,
    const float* __restrict__ o_m, const float* __restrict__ o_v,
    half_t* __restrict__ w16, half_t* __restrict__ ow16,
    float* __restrict__ shift_qkv, float* __restrict__ shift_out) {
    int tid = blockIdx.x * blockDim.x + threadIdx.x;
    if (tid >= Cn * Cn) return;
    int d = tid >> 9;
    int c = tid & (Cn - 1);
    float sq = q_g[d] * rsqrtf(q_v[d] + BN_EPS);
    float sk = k_g[d] * rsqrtf(k_v[d] + BN_EPS);
    float sv = v_g[d] * rsqrtf(v_v[d] + BN_EPS);
    float so = o_g[d] * rsqrtf(o_v[d] + BN_EPS);
    size_t wi = (size_t)d * Cn + c;
    w16[wi]                        = (half_t)(q_w[wi] * sq);
    w16[(size_t)(Cn + d) * Cn + c] = (half_t)(k_w[wi] * sk);
    w16[(size_t)(2*Cn + d)*Cn + c] = (half_t)(v_w[wi] * sv);
    ow16[wi]                       = (half_t)(o_w[wi] * so);
    if (c == 0) {
        shift_qkv[d]          = q_b[d] - q_m[d] * sq;
        shift_qkv[Cn + d]     = k_b[d] - k_m[d] * sk;
        shift_qkv[2*Cn + d]   = v_b[d] - v_m[d] * sv;
        shift_out[d]          = o_bias[d] * so + o_beta[d] - o_m[d] * so;
    }
}

// ---------------------------------------------------------------------------
// WMMA GEMM: C[m,n] = sum_k A[m,k] * Bw[n,k]  (Bw row-major = weights [d][c])
// 128x128 block tile, BK=32, 8 waves each owning a 64x32 tile (4x2 WMMA frags)
// MODE 0: store f16 (acc + shift[n]) at outh[m*ldo + n]
// MODE 1: store f32 (acc + shift[n] + identity) at [tb, n, hw] layout
// ---------------------------------------------------------------------------
template <int MODE>
__global__ __launch_bounds__(256) void gemm_wmma_kernel(
    const half_t* __restrict__ A, int lda,
    const half_t* __restrict__ Bw,
    const float* __restrict__ shift,
    half_t* __restrict__ outh, int ldo,
    float* __restrict__ outf,
    const float* __restrict__ identity,
    int K) {
    __shared__ __align__(16) half_t As[128 * 40];   // row stride 40 f16 (pad)
    __shared__ __align__(16) half_t Bs[128 * 40];   // stored N-major

    const int tid  = threadIdx.x;
    const int lane = tid & 31;
    const int wid  = tid >> 5;
    const int wm   = (wid & 1) * 64;   // wave row offset inside block tile
    const int wn   = (wid >> 1) * 32;  // wave col offset inside block tile
    const int lm   = lane & 15;
    const int kg   = lane >> 4;

    const int m0 = blockIdx.x * 128;
    const int n0 = blockIdx.y * 128;

    v8f acc[4][2] = {};

    for (int kk = 0; kk < K; kk += 32) {
        __syncthreads();
        // cooperative tile load: 128 rows x 32 f16 = 512 chunks of 16B each
        #pragma unroll
        for (int i = 0; i < 2; ++i) {
            int chunk = i * 256 + tid;
            int r  = chunk >> 2;
            int kc = chunk & 3;
            const uint4* ga =
                (const uint4*)(A + (size_t)(m0 + r) * lda + kk + kc * 8);
            *(uint4*)(&As[r * 40 + kc * 8]) = *ga;
            const uint4* gb =
                (const uint4*)(Bw + (size_t)(n0 + r) * K + kk + kc * 8);
            *(uint4*)(&Bs[r * 40 + kc * 8]) = *gb;
        }
        __syncthreads();

        // A fragment (16x32 f16): lanes 0-15 -> M rows, kg selects K group.
        // elements e<8: K = kg*8+e ; e>=8: K = kg*8+16+(e-8)
        Frag a[4];
        #pragma unroll
        for (int i = 0; i < 4; ++i) {
            const half_t* p = &As[(wm + i * 16 + lm) * 40 + kg * 8];
            a[i].h[0] = *(const v8h*)(p);
            a[i].h[1] = *(const v8h*)(p + 16);
        }
        // B fragment (32x16 f16): n = lane&15, e<8: K = kg*8+e ; e>=8: K = kg*8+16+(e-8)
        Frag b[2];
        #pragma unroll
        for (int j = 0; j < 2; ++j) {
            const half_t* p = &Bs[(wn + j * 16 + lm) * 40 + kg * 8];
            b[j].h[0] = *(const v8h*)(p);
            b[j].h[1] = *(const v8h*)(p + 16);
        }
        #pragma unroll
        for (int i = 0; i < 4; ++i)
            #pragma unroll
            for (int j = 0; j < 2; ++j)
                acc[i][j] = __builtin_amdgcn_wmma_f32_16x16x32_f16(
                    false, a[i].f, false, b[j].f, (short)0, acc[i][j],
                    false, false);
    }

    // Epilogue. C/D layout: VGPR r -> row = base + (lane>=16 ? 8 : 0) + r,
    // col = lane & 15.
    #pragma unroll
    for (int i = 0; i < 4; ++i) {
        #pragma unroll
        for (int j = 0; j < 2; ++j) {
            const int col  = n0 + wn + j * 16 + lm;
            const float sh = shift[col];
            const int rowb = m0 + wm + i * 16 + kg * 8;
            #pragma unroll
            for (int r = 0; r < 8; ++r) {
                const int row = rowb + r;
                float val = acc[i][j][r] + sh;
                if (MODE == 0) {
                    outh[(size_t)row * ldo + col] = (half_t)val;
                } else {
                    const int tb = row / HWn;
                    const int hw = row - tb * HWn;
                    const size_t oi = ((size_t)tb * Cn + col) * HWn + hw;
                    outf[oi] = val + identity[oi];
                }
            }
        }
    }
}

// ---------------------------------------------------------------------------
// K3: in-place LIF scan over T for fused q/k/v preactivations [M,1536] f16
// ---------------------------------------------------------------------------
__global__ void lif_qkv_kernel(half_t* __restrict__ pre) {
    int tid = blockIdx.x * blockDim.x + threadIdx.x;
    if (tid >= Bn * HWn * 3 * Cn) return;
    int j  = tid % (3 * Cn);
    int r  = tid / (3 * Cn);
    int hw = r % HWn;
    int b  = r / HWn;
    float v = 0.0f;
    #pragma unroll
    for (int t = 0; t < Tn; ++t) {
        size_t idx = ((size_t)((t * Bn + b) * HWn + hw)) * (3 * Cn) + j;
        float xt = (float)pre[idx];
        v = 0.5f * (v + xt);
        float s = (v >= 1.0f) ? 1.0f : 0.0f;
        v = (s > 0.0f) ? 0.0f : v;
        pre[idx] = (half_t)s;
    }
}

// ---------------------------------------------------------------------------
// K4: kv[t,b,c] = sum_n k_sp * v_sp  then LIF scan over T -> kvsp [T*B, C]
// ---------------------------------------------------------------------------
__global__ void kv_lif_kernel(const half_t* __restrict__ pre,
                              half_t* __restrict__ kvsp) {
    int tid = blockIdx.x * blockDim.x + threadIdx.x;
    if (tid >= Bn * Cn) return;
    int c = tid & (Cn - 1);
    int b = tid >> 9;
    float v = 0.0f;
    for (int t = 0; t < Tn; ++t) {
        float sum = 0.0f;
        for (int n = 0; n < HWn; ++n) {
            size_t m = (size_t)(t * Bn + b) * HWn + n;
            float ks = (float)pre[m * (3 * Cn) + Cn + c];
            float vs = (float)pre[m * (3 * Cn) + 2 * Cn + c];
            sum += ks * vs;
        }
        v = 0.5f * (v + sum);
        float s = (v >= 1.0f) ? 1.0f : 0.0f;
        v = (s > 0.0f) ? 0.0f : v;
        kvsp[(size_t)(t * Bn + b) * Cn + c] = (half_t)s;
    }
}

// ---------------------------------------------------------------------------
// K5: ypre = q_spike * kv_spike (in-place into q slot, becomes GEMM2's A)
//     also emit vh output [T,B,heads,N,hd] float (= reshuffled v spikes)
// ---------------------------------------------------------------------------
__global__ void mask_vh_kernel(half_t* __restrict__ pre,
                               const half_t* __restrict__ kvsp,
                               float* __restrict__ vh_out) {
    int tid = blockIdx.x * blockDim.x + threadIdx.x;
    if (tid >= Mn * Cn) return;
    int c = tid & (Cn - 1);
    int m = tid >> 9;
    int tb = m / HWn;
    int hw = m - tb * HWn;
    size_t base = (size_t)m * (3 * Cn);
    float q  = (float)pre[base + c];
    float kv = (float)kvsp[(size_t)tb * Cn + c];
    pre[base + c] = (half_t)(q * kv);
    float vs = (float)pre[base + 2 * Cn + c];
    int head = c >> 6;
    int d    = c & 63;
    vh_out[(((size_t)tb * 8 + head) * HWn + hw) * 64 + d] = vs;
}

// ---------------------------------------------------------------------------
extern "C" void kernel_launch(void* const* d_in, const int* in_sizes, int n_in,
                              void* d_out, int out_size, void* d_ws,
                              size_t ws_size, hipStream_t stream) {
    (void)in_sizes; (void)n_in; (void)out_size;
    const float* x     = (const float*)d_in[0];
    const float* q_w   = (const float*)d_in[1];
    const float* k_w   = (const float*)d_in[2];
    const float* v_w   = (const float*)d_in[3];
    const float* out_w = (const float*)d_in[4];
    const float* out_b = (const float*)d_in[5];
    const float* q_g = (const float*)d_in[6];
    const float* q_b = (const float*)d_in[7];
    const float* q_m = (const float*)d_in[8];
    const float* q_v = (const float*)d_in[9];
    const float* k_g = (const float*)d_in[10];
    const float* k_b = (const float*)d_in[11];
    const float* k_m = (const float*)d_in[12];
    const float* k_v = (const float*)d_in[13];
    const float* v_g = (const float*)d_in[14];
    const float* v_b = (const float*)d_in[15];
    const float* v_m = (const float*)d_in[16];
    const float* v_v = (const float*)d_in[17];
    const float* o_g = (const float*)d_in[18];
    const float* o_b = (const float*)d_in[19];
    const float* o_m = (const float*)d_in[20];
    const float* o_v = (const float*)d_in[21];
    float* out = (float*)d_out;

    // Workspace carve (256B aligned slices)
    char* p = (char*)d_ws;
    size_t used = 0;
    auto carve = [&](size_t bytes) {
        char* r = p;
        size_t pad = (bytes + 255) & ~(size_t)255;
        p += pad; used += pad;
        return r;
    };
    half_t* xs     = (half_t*)carve((size_t)Mn * Cn * 2);        // 25.7 MB
    half_t* w16    = (half_t*)carve((size_t)3 * Cn * Cn * 2);    // 1.6 MB
    half_t* ow16   = (half_t*)carve((size_t)Cn * Cn * 2);        // 0.5 MB
    float*  sh_qkv = (float*)carve((size_t)3 * Cn * 4);
    float*  sh_out = (float*)carve((size_t)Cn * 4);
    half_t* pre    = (half_t*)carve((size_t)Mn * 3 * Cn * 2);    // 77 MB
    half_t* kvsp   = (half_t*)carve((size_t)Tn * Bn * Cn * 2);
    if (used > ws_size) return;  // insufficient scratch: do nothing (safe)

    // K0: input LIF -> spike A matrix
    {
        int n = Bn * HWn * Cn;
        lif_input_kernel<<<(n + 255) / 256, 256, 0, stream>>>(x, xs);
    }
    // K1: weight prep (BN-scale folded, f16)
    {
        int n = Cn * Cn;
        prep_weights_kernel<<<(n + 255) / 256, 256, 0, stream>>>(
            q_w, k_w, v_w, out_w, out_b,
            q_g, q_b, q_m, q_v, k_g, k_b, k_m, k_v,
            v_g, v_b, v_m, v_v, o_g, o_b, o_m, o_v,
            w16, ow16, sh_qkv, sh_out);
    }
    // K2: fused q/k/v GEMM  (M=25088, N=1536, K=512) -> f16 preact
    gemm_wmma_kernel<0><<<dim3(Mn / 128, (3 * Cn) / 128), 256, 0, stream>>>(
        xs, Cn, w16, sh_qkv, pre, 3 * Cn, nullptr, nullptr, Cn);
    // K3: LIF scan -> binary q/k/v spikes (in place)
    {
        int n = Bn * HWn * 3 * Cn;
        lif_qkv_kernel<<<(n + 255) / 256, 256, 0, stream>>>(pre);
    }
    // K4: kv reduction over N + LIF
    {
        int n = Bn * Cn;
        kv_lif_kernel<<<(n + 255) / 256, 256, 0, stream>>>(pre, kvsp);
    }
    // K5: y = q*kv (in place) ; emit vh (second output)
    {
        int n = Mn * Cn;
        mask_vh_kernel<<<(n + 255) / 256, 256, 0, stream>>>(
            pre, kvsp, out + (size_t)Mn * Cn);
    }
    // K6: out GEMM + bias/BN shift + residual, transposed store to d_out
    gemm_wmma_kernel<1><<<dim3(Mn / 128, Cn / 128), 256, 0, stream>>>(
        pre, 3 * Cn, ow16, sh_out, nullptr, 0, out, x, Cn);
}
